// Attention_12970801234663
// MI455X (gfx1250) — matrix operand order repaired and checked by reading, not compiled, not run on hardware
//
#include <hip/hip_runtime.h>
#include <math.h>

// ---------------------------------------------------------------------------
// CDNA5 (gfx1250, wave32) attention block:
//   v_wmma_f32_16x16x32_f16 for all GEMMs, tensor_load_to_lds (TDM) staging
//   of K/V per workgroup, s_wait_tensorcnt, flash softmax on half-wave shfl.
// ---------------------------------------------------------------------------

typedef _Float16 h16;
typedef __attribute__((ext_vector_type(16))) _Float16 v16h;
typedef __attribute__((ext_vector_type(8)))  _Float16 v8h;
typedef __attribute__((ext_vector_type(8)))  float    v8f;
typedef __attribute__((ext_vector_type(4)))  unsigned int u32x4;
typedef __attribute__((ext_vector_type(8)))  int      i32x8;
typedef __attribute__((ext_vector_type(4)))  int      i32x4;

static __device__ __forceinline__ v8f wmma_f16(v16h a, v16h b, v8f c) {
  // D = A(16x32 f16) * B(32x16 f16) + C(16x16 f32)
  return __builtin_amdgcn_wmma_f32_16x16x32_f16(false, a, false, b, (short)0, c,
                                                false, false);
}

static __device__ __forceinline__ v16h cat8(v8h lo, v8h hi) {
  return __builtin_shufflevector(lo, hi, 0, 1, 2, 3, 4, 5, 6, 7,
                                          8, 9, 10, 11, 12, 13, 14, 15);
}

// 16-lane (half-wave) reductions: rows of a WMMA C tile live across 16 lanes.
static __device__ __forceinline__ float hmax16(float x) {
  x = fmaxf(x, __shfl_xor(x, 1, 16));
  x = fmaxf(x, __shfl_xor(x, 2, 16));
  x = fmaxf(x, __shfl_xor(x, 4, 16));
  x = fmaxf(x, __shfl_xor(x, 8, 16));
  return x;
}
static __device__ __forceinline__ float hsum16(float x) {
  x += __shfl_xor(x, 1, 16);
  x += __shfl_xor(x, 2, 16);
  x += __shfl_xor(x, 4, 16);
  x += __shfl_xor(x, 8, 16);
  return x;
}

#define B_     8
#define C_     256
#define HW_    1024
#define NH_    8
#define HD_    32

// ---------------------------------------------------------------------------
// Tensor Data Mover: 64KB contiguous global -> LDS copy as a 1-D tensor of
// 8192 x 8-byte elements. D# packing per CDNA5 ISA 8.3/8.4 (groups 0 and 1;
// groups 2/3 zero => <=2D). Completion tracked by TENSORcnt.
// ---------------------------------------------------------------------------
static __device__ __forceinline__ void tdm_load_64k(const void* src,
                                                    unsigned lds_off) {
  unsigned long long ga = (unsigned long long)(size_t)src;
  u32x4 g0;
  g0[0] = 1u;                                        // count=1 (valid)
  g0[1] = lds_off;                                   // lds_addr (bytes)
  g0[2] = (unsigned)(ga & 0xffffffffu);              // global_addr[31:0]
  g0[3] = (unsigned)((ga >> 32) & 0x1ffffffu)        // global_addr[56:32]
        | (2u << 30);                                // type=2 ("image")
  i32x8 g1;
  g1[0] = (int)(3u << 16);          // wg_mask=0, data_size=3 (8 bytes)
  g1[1] = (int)(8192u << 16);       // tensor_dim0[15:0]=8192
  g1[2] = (int)(1u << 16);          // tensor_dim0[31:16]=0, tensor_dim1=1
  g1[3] = (int)(8192u << 16);       // tile_dim0=8192 (1-D tile, 64KB)
  g1[4] = 0;                        // tile_dim1=0, tile_dim2=0 (unused)
  g1[5] = (int)8192;                // tensor_dim0_stride[31:0]
  g1[6] = 0;
  g1[7] = 0;
  i32x4 z4 = {0, 0, 0, 0};
#if defined(__clang_major__) && (__clang_major__ >= 23)
  i32x8 z8 = {0, 0, 0, 0, 0, 0, 0, 0};
  __builtin_amdgcn_tensor_load_to_lds(g0, g1, z4, z4, z8, 0);
#else
  __builtin_amdgcn_tensor_load_to_lds(g0, g1, z4, z4, 0);
#endif
}

// ---------------------------------------------------------------------------
// K1: convert weights to f16; fold softmax scale * log2(e) into q rows/bias.
// ---------------------------------------------------------------------------
__global__ void k_convert(const float* __restrict__ qkv_w,
                          const float* __restrict__ qkv_b,
                          const float* __restrict__ out_w,
                          h16* __restrict__ wqkv, float* __restrict__ bqkv,
                          h16* __restrict__ wout) {
  const float QS = 0.17677669529663687f * 1.4426950408889634f; // 32^-0.5*log2e
  int i = blockIdx.x * 256 + threadIdx.x;
  int stride = gridDim.x * 256;
  for (int t = i; t < 3 * C_ * C_; t += stride) {
    float s = (t < C_ * C_) ? QS : 1.0f;
    wqkv[t] = (h16)(qkv_w[t] * s);
  }
  for (int t = i; t < C_ * C_; t += stride) wout[t] = (h16)out_w[t];
  for (int t = i; t < 3 * C_; t += stride)
    bqkv[t] = qkv_b[t] * ((t < C_) ? QS : 1.0f);
}

// ---------------------------------------------------------------------------
// K2: GroupNorm(32 groups). One block per (batch, group); writes xnT[b][p][c]
// in f16 so the QKV GEMM's B-operand loads are contiguous along K (=c).
// ---------------------------------------------------------------------------
__global__ void k_gnorm(const float* __restrict__ x, const float* __restrict__ gw,
                        const float* __restrict__ gb, h16* __restrict__ xnT) {
  __shared__ float r0[256];
  __shared__ float r1[256];
  int bb = blockIdx.x >> 5, g = blockIdx.x & 31, c0 = g * 8;
  const float* xb = x + ((size_t)bb * C_ + c0) * HW_;   // 8 channels contiguous
  float s = 0.f, sq = 0.f;
  for (int t = threadIdx.x; t < 8 * HW_; t += 256) {
    float v = xb[t];
    s += v; sq += v * v;
  }
  r0[threadIdx.x] = s; r1[threadIdx.x] = sq;
  __syncthreads();
  for (int off = 128; off > 0; off >>= 1) {
    if ((int)threadIdx.x < off) {
      r0[threadIdx.x] += r0[threadIdx.x + off];
      r1[threadIdx.x] += r1[threadIdx.x + off];
    }
    __syncthreads();
  }
  float mu = r0[0] * (1.0f / 8192.0f);
  float var = r1[0] * (1.0f / 8192.0f) - mu * mu;
  float rs = rsqrtf(var + 1e-5f);
  h16* ob = xnT + (size_t)bb * HW_ * C_ + c0;
  for (int t = threadIdx.x; t < 8 * HW_; t += 256) {
    int c = t >> 10, p = t & (HW_ - 1);
    float v = (xb[t] - mu) * rs * gw[c0 + c] + gb[c0 + c];
    ob[(size_t)p * C_ + c] = (h16)v;
  }
}

// ---------------------------------------------------------------------------
// K3: QKV GEMM per batch: [768x256] x [256x1024]. One wave -> 16x64 strip,
// register double-buffered over K so loads run one WMMA group ahead.
// Scatters into WMMA-friendly layouts: qT/kT[bh][p][d], v[bh][d][p].
// ---------------------------------------------------------------------------
__global__ void __launch_bounds__(256)
k_qkv(const h16* __restrict__ W, const float* __restrict__ bias,
      const h16* __restrict__ xnT, h16* __restrict__ qT,
      h16* __restrict__ kT, h16* __restrict__ vv) {
  int lane = threadIdx.x & 31;
  int job = blockIdx.x * 8 + (threadIdx.x >> 5);  // 6144 wave-jobs
  int b = job / 768, rem = job % 768;
  int m0 = (rem >> 4) * 16, n0 = (rem & 15) * 64;
  int ksel = lane >> 4, ln = lane & 15;

  const h16* Arow = W + (size_t)(m0 + ln) * C_;
  const h16* Bb = xnT + (size_t)b * HW_ * C_;
  v8f acc[4] = {};
  v16h A = *(const v16h*)(Arow + ksel * 16);
  v16h Bt[4];
#pragma unroll
  for (int nt = 0; nt < 4; nt++)
    Bt[nt] = *(const v16h*)(Bb + (size_t)(n0 + nt * 16 + ln) * C_ + ksel * 16);

#pragma unroll
  for (int k0 = 32; k0 <= C_; k0 += 32) {
    v16h An = A;
    v16h Bn[4] = {Bt[0], Bt[1], Bt[2], Bt[3]};
    if (k0 < C_) {  // prefetch next K-step before consuming current
      An = *(const v16h*)(Arow + k0 + ksel * 16);
#pragma unroll
      for (int nt = 0; nt < 4; nt++)
        Bn[nt] = *(const v16h*)(Bb + (size_t)(n0 + nt * 16 + ln) * C_ + k0 +
                                ksel * 16);
    }
#pragma unroll
    for (int nt = 0; nt < 4; nt++) acc[nt] = wmma_f16(A, Bt[nt], acc[nt]);
    A = An;
#pragma unroll
    for (int nt = 0; nt < 4; nt++) Bt[nt] = Bn[nt];
  }

  int hi = (lane < 16) ? 0 : 8;
#pragma unroll
  for (int nt = 0; nt < 4; nt++) {
    int p = n0 + nt * 16 + ln;
#pragma unroll
    for (int r = 0; r < 8; r++) {
      int o = m0 + r + hi;
      float val = acc[nt][r] + bias[o];
      if (o < C_) {
        int head = o >> 5, d = o & 31;
        qT[(((size_t)b * NH_ + head) * HW_ + p) * HD_ + d] = (h16)val;
      } else if (o < 2 * C_) {
        int oo = o - C_, head = oo >> 5, d = oo & 31;
        kT[(((size_t)b * NH_ + head) * HW_ + p) * HD_ + d] = (h16)val;
      } else {
        int oo = o - 2 * C_, head = oo >> 5, d = oo & 31;
        vv[(((size_t)b * NH_ + head) * HD_ + d) * HW_ + p] = (h16)val;
      }
    }
  }
}

// ---------------------------------------------------------------------------
// K4: Flash attention. All 8 waves of a block share one (b,head); K^T and V
// for that head (64KB each, contiguous) are staged into LDS once by the
// Tensor Data Mover, then consumed 32x per wave as ds_load_b128 pairs.
// One wave owns one 16-row query tile; K=32==head_dim -> one WMMA per sim
// tile. Online softmax stats are per-lane (C layout gives each lane 8 rows).
// Probs bounce through a 1KB/wave LDS tile to become the next A operand.
// Epilogue writes the reference's raw-reshape layout transposed:
// inp2T[b][p'][c], p'=(i&31)*32+d, c=h*32+(i>>5).
// ---------------------------------------------------------------------------
#define KV_BYTES (2 * HW_ * HD_ * 2)          // 128KB: K then V
__global__ void __launch_bounds__(256)
k_attn(const h16* __restrict__ qT, const h16* __restrict__ kT,
       const h16* __restrict__ vv, h16* __restrict__ inp2T) {
  __shared__ __align__(16) unsigned char smem[KV_BYTES + 8 * 16 * 40 * 2];
  int lane = threadIdx.x & 31, wid = threadIdx.x >> 5;
  int bh = blockIdx.x >> 3;                   // 8 blocks per (b,head)
  int b = bh >> 3, hh = bh & 7;
  int i0 = ((blockIdx.x & 7) * 8 + wid) * 16; // this wave's query tile
  int ksel = lane >> 4, ln = lane & 15;

  // --- TDM staging of K^T[bh] and V[bh] into LDS (wave 0 issues) ---
  if (threadIdx.x < 32) {
    unsigned lds0 = (unsigned)(size_t)(void*)smem;  // flat addr low 32 = LDS
    tdm_load_64k(kT + (size_t)bh * HW_ * HD_, lds0);
    tdm_load_64k(vv + (size_t)bh * HD_ * HW_, lds0 + HW_ * HD_ * 2);
    __builtin_amdgcn_s_wait_tensorcnt(0);
  }
  __syncthreads();
  const h16* kl = (const h16*)smem;                    // [p][d]
  const h16* vl = (const h16*)(smem + HW_ * HD_ * 2);  // [d][p]
  h16* lptr = (h16*)(smem + KV_BYTES) + wid * 16 * 40;

  const h16* qbase = qT + ((size_t)bh * HW_ + i0 + ln) * HD_;
  v16h Aq = *(const v16h*)(qbase + ksel * 16);

  float m[8], l[8];
#pragma unroll
  for (int r = 0; r < 8; r++) { m[r] = -3.0e38f; l[r] = 0.f; }
  v8f acc0{}, acc1{}, z{};

  for (int j0 = 0; j0 < HW_; j0 += 32) {
    v16h Bk0 = *(const v16h*)(kl + (size_t)(j0 + ln) * HD_ + ksel * 16);
    v16h Bk1 = *(const v16h*)(kl + (size_t)(j0 + 16 + ln) * HD_ + ksel * 16);
    v8f s0 = wmma_f16(Aq, Bk0, z);
    v8f s1 = wmma_f16(Aq, Bk1, z);
    v8f p0, p1;
#pragma unroll
    for (int r = 0; r < 8; r++) {
      float t = hmax16(fmaxf(s0[r], s1[r]));
      float mn = fmaxf(m[r], t);
      float al = exp2f(m[r] - mn);          // log2e folded into q scale
      float e0 = exp2f(s0[r] - mn);
      float e1 = exp2f(s1[r] - mn);
      float rs = hsum16(e0 + e1);
      l[r] = l[r] * al + rs;
      m[r] = mn;
      acc0[r] *= al; acc1[r] *= al;
      p0[r] = e0; p1[r] = e1;
    }
    // C-layout -> A-layout relayout via per-wave LDS tile
#pragma unroll
    for (int r = 0; r < 8; r++) {
      int row = r + ((lane < 16) ? 0 : 8);
      lptr[row * 40 + ln]      = (h16)p0[r];
      lptr[row * 40 + 16 + ln] = (h16)p1[r];
    }
    __syncthreads();  // uniform trip count across all 8 waves
    const h16* rp = lptr + ln * 40 + ksel * 16;
    v16h Ap = cat8(*(const v8h*)rp, *(const v8h*)(rp + 8));
    v16h Bv0 = *(const v16h*)(vl + (size_t)ln * HW_ + j0 + ksel * 16);
    v16h Bv1 = *(const v16h*)(vl + (size_t)(16 + ln) * HW_ + j0 + ksel * 16);
    acc0 = wmma_f16(Ap, Bv0, acc0);
    acc1 = wmma_f16(Ap, Bv1, acc1);
  }
  h16* ob = inp2T + (size_t)b * HW_ * C_;
#pragma unroll
  for (int r = 0; r < 8; r++) {
    float inv = 1.0f / l[r];
    int gi = i0 + r + ((lane < 16) ? 0 : 8);
    int c = hh * 32 + (gi >> 5);
    int pb = (gi & 31) * 32;
    ob[(size_t)(pb + ln) * C_ + c]      = (h16)(acc0[r] * inv);
    ob[(size_t)(pb + 16 + ln) * C_ + c] = (h16)(acc1[r] * inv);
  }
}

// ---------------------------------------------------------------------------
// K5: out projection [256x256]x[256x1024] + bias + residual -> f32 output.
// Register double-buffered over K like K3.
// ---------------------------------------------------------------------------
__global__ void __launch_bounds__(256)
k_out(const h16* __restrict__ W, const float* __restrict__ bias,
      const h16* __restrict__ inp2T, const float* __restrict__ x,
      float* __restrict__ y) {
  int lane = threadIdx.x & 31;
  int job = blockIdx.x * 8 + (threadIdx.x >> 5);  // 2048 wave-jobs
  int b = job >> 8, rem = job & 255;
  int m0 = (rem >> 4) * 16, n0 = (rem & 15) * 64;
  int ksel = lane >> 4, ln = lane & 15;
  const h16* Arow = W + (size_t)(m0 + ln) * C_;
  const h16* Bb = inp2T + (size_t)b * HW_ * C_;
  v8f acc[4] = {};
  v16h A = *(const v16h*)(Arow + ksel * 16);
  v16h Bt[4];
#pragma unroll
  for (int nt = 0; nt < 4; nt++)
    Bt[nt] = *(const v16h*)(Bb + (size_t)(n0 + nt * 16 + ln) * C_ + ksel * 16);

#pragma unroll
  for (int k0 = 32; k0 <= C_; k0 += 32) {
    v16h An = A;
    v16h Bn[4] = {Bt[0], Bt[1], Bt[2], Bt[3]};
    if (k0 < C_) {
      An = *(const v16h*)(Arow + k0 + ksel * 16);
#pragma unroll
      for (int nt = 0; nt < 4; nt++)
        Bn[nt] = *(const v16h*)(Bb + (size_t)(n0 + nt * 16 + ln) * C_ + k0 +
                                ksel * 16);
    }
#pragma unroll
    for (int nt = 0; nt < 4; nt++) acc[nt] = wmma_f16(A, Bt[nt], acc[nt]);
    A = An;
#pragma unroll
    for (int nt = 0; nt < 4; nt++) Bt[nt] = Bn[nt];
  }

  int hi = (lane < 16) ? 0 : 8;
#pragma unroll
  for (int nt = 0; nt < 4; nt++) {
    int p = n0 + nt * 16 + ln;
#pragma unroll
    for (int r = 0; r < 8; r++) {
      int o = m0 + r + hi;
      size_t idx = ((size_t)b * C_ + o) * HW_ + p;
      y[idx] = acc[nt][r] + bias[o] + x[idx];
    }
  }
}

// ---------------------------------------------------------------------------
extern "C" void kernel_launch(void* const* d_in, const int* in_sizes, int n_in,
                              void* d_out, int out_size, void* d_ws,
                              size_t ws_size, hipStream_t stream) {
  (void)in_sizes; (void)n_in; (void)out_size; (void)ws_size;
  const float* x      = (const float*)d_in[0];
  const float* norm_w = (const float*)d_in[1];
  const float* norm_b = (const float*)d_in[2];
  const float* qkv_w  = (const float*)d_in[3];
  const float* qkv_b  = (const float*)d_in[4];
  const float* out_w  = (const float*)d_in[5];
  const float* out_b  = (const float*)d_in[6];
  float* y = (float*)d_out;

  char* ws = (char*)d_ws;
  auto alloc = [&](size_t bytes) -> char* {
    char* p = ws;
    ws += (bytes + 255) & ~(size_t)255;
    return p;
  };
  h16*   wqkv = (h16*)  alloc((size_t)3 * C_ * C_ * 2);
  h16*   wout = (h16*)  alloc((size_t)C_ * C_ * 2);
  float* bqkv = (float*)alloc((size_t)3 * C_ * 4);
  h16*   xnT  = (h16*)  alloc((size_t)B_ * HW_ * C_ * 2);
  h16*   qT   = (h16*)  alloc((size_t)B_ * NH_ * HW_ * HD_ * 2);
  h16*   kT   = (h16*)  alloc((size_t)B_ * NH_ * HW_ * HD_ * 2);
  h16*   vv   = (h16*)  alloc((size_t)B_ * NH_ * HD_ * HW_ * 2);
  h16*   i2T  = (h16*)  alloc((size_t)B_ * HW_ * C_ * 2);

  k_convert<<<64, 256, 0, stream>>>(qkv_w, qkv_b, out_w, wqkv, bqkv, wout);
  k_gnorm  <<<B_ * 32, 256, 0, stream>>>(x, norm_w, norm_b, xnT);
  k_qkv    <<<768, 256, 0, stream>>>(wqkv, bqkv, xnT, qT, kT, vv);
  k_attn   <<<512, 256, 0, stream>>>(qT, kT, vv, i2T);
  k_out    <<<256, 256, 0, stream>>>(wout, out_b, i2T, x, y);
}
